// Conv4d_groups_72937134621191
// MI455X (gfx1250) — matrix operand order repaired and checked by reading, not compiled, not run on hardware
//
#include <hip/hip_runtime.h>

// ---------------------------------------------------------------------------
// 4D circular conv (K=3^4) on MI455X / gfx1250, wave32 WMMA path.
//   out[b,co,t,y,z,w] = bias[co]
//     + sum_{dt,dy,dz,dw in {-1,0,1}} sum_ci
//         x[b,ci,(t+dt)%16,(y+dy)%16,(z+dz)%16,(w+dw)%16] * W[dt+1,co,ci,dy+1,dz+1,dw+1]
// Implemented as 81 taps of a (c_out=64) x (c_in=64) GEMM, N = 16 w-positions
// (dw taps are rotations along N), z-tiled by 2. bf16 hi/lo split with f32
// accumulation on v_wmma_f32_16x16x32_bf16.
// ---------------------------------------------------------------------------

typedef __bf16 bf16;
typedef __attribute__((ext_vector_type(8)))  bf16  v8bf;
typedef __attribute__((ext_vector_type(16))) bf16  v16bf;
typedef __attribute__((ext_vector_type(8)))  float v8f;

#define S        16
#define CIN      64
#define COUT     64
#define NTAP81   81
#define APK_ELEMS (NTAP81 * 2 * COUT * 32)      // 331776 bf16 per (hi|lo) plane
#define ROWS36   36                              // (3 dt * 3 dy) * 4 z-input rows
#define LDS_HALF (ROWS36 * 2 * 16 * 32)          // 36864 bf16 elems (hi or lo)

// --- weight pre-pack: W[it][co][ci][iy][iz][iw] f32 -> A-fragment bf16 hi/lo
//     layout Apk[t81][cb][co][k32], so an A frag is two contiguous 16B loads.
__global__ void pack_w_kernel(const float* __restrict__ w,
                              bf16* __restrict__ ahi, bf16* __restrict__ alo) {
  int e = blockIdx.x * 256 + threadIdx.x;
  if (e >= APK_ELEMS) return;
  int k   = e & 31;
  int co  = (e >> 5) & 63;
  int cb  = (e >> 11) & 1;
  int t81 = e >> 12;
  int iw = t81 % 3, iz = (t81 / 3) % 3, iy = (t81 / 9) % 3, it = t81 / 27;
  int ci = cb * 32 + k;
  size_t src = (((((size_t)it * COUT + co) * CIN + ci) * 3 + iy) * 3 + iz) * 3 + iw;
  float v = w[src];
  bf16 h = (bf16)v;                 // round-to-nearest-even high part
  bf16 l = (bf16)(v - (float)h);    // residual low part
  ahi[e] = h;
  alo[e] = l;
}

// A-fragment loader: lane<16 needs K {0..7, 16..23}, lane>=16 K {8..15, 24..31};
// with aoff = (...)*32 + (lane>>4)*8 this is 16B at +0 and 16B at +32 bytes.
static __device__ __forceinline__ v16bf ld_afrag(const bf16* p) {
  v8bf lo = *(const v8bf*)(p);
  v8bf hi = *(const v8bf*)(p + 16);
  return __builtin_shufflevector(lo, hi, 0, 1, 2, 3, 4, 5, 6, 7,
                                 8, 9, 10, 11, 12, 13, 14, 15);
}

__global__ __launch_bounds__(128, 1) void conv4d_wmma_kernel(
    const float* __restrict__ x,
    const bf16* __restrict__ ahi, const bf16* __restrict__ alo,
    const float* __restrict__ bias, float* __restrict__ out) {
  extern __shared__ bf16 lds[];            // [hi|lo][row36][cb][w16][k32]
  bf16* bhiS = lds;
  bf16* bloS = lds + LDS_HALF;

  const int blk = blockIdx.x;              // 4*16*16*8 = 8192 blocks
  const int z0 = (blk & 7) * 2;
  const int y  = (blk >> 3) & 15;
  const int t  = (blk >> 7) & 15;
  const int b  = blk >> 11;

  const int tid  = threadIdx.x;
  const int lane = tid & 31;               // wave32
  const int wave = tid >> 5;               // c_out block 0..3

  // ---- stage input patch to LDS in bf16 hi/lo, B-fragment layout ----------
  // rows: (it,iy) in 3x3, zi in 0..3 (input z = z0-1..z0+2), ci in 0..63
  for (int r = tid; r < 9 * 4 * CIN; r += 128) {
    int ci = r & 63;
    int zi = (r >> 6) & 3;
    int q  = r >> 8;                       // 0..8 = it*3+iy
    int iy = q % 3, it = q / 3;
    int T = (t + it + 15) & 15;
    int Y = (y + iy + 15) & 15;
    int Z = (z0 + zi + 15) & 15;
    const float4* src =
        (const float4*)(x + ((((((size_t)b * CIN + ci) * S + T) * S + Y) * S + Z) << 4));
    float4 f0 = src[0], f1 = src[1], f2 = src[2], f3 = src[3];
    float vals[16] = {f0.x, f0.y, f0.z, f0.w, f1.x, f1.y, f1.z, f1.w,
                      f2.x, f2.y, f2.z, f2.w, f3.x, f3.y, f3.z, f3.w};
    int base = (((q * 4 + zi) * 2 + (ci >> 5)) << 9) + (ci & 31);  // [row][cb][w][k]
#pragma unroll
    for (int w = 0; w < 16; ++w) {
      float v = vals[w];
      bf16 h = (bf16)v;
      bhiS[base + w * 32] = h;
      bloS[base + w * 32] = (bf16)(v - (float)h);
    }
  }
  __syncthreads();

  // ---- K loop: 81 taps x 2 ci-chunks; 2 N-tiles (z0, z0+1); 3 split WMMAs --
  v8f acc0 = {};
  v8f acc1 = {};
  const int aoff = ((wave << 4) + (lane & 15)) * 32 + ((lane >> 4) << 3);
  const int bk   = (lane >> 4) << 3;       // lanes>=16 read K 8..15, 24..31
  const int wl   = lane & 15;              // N position (w)

#pragma unroll 1
  for (int it = 0; it < 3; ++it)
#pragma unroll 1
    for (int iy = 0; iy < 3; ++iy)
#pragma unroll 1
      for (int idz = 0; idz < 3; ++idz) {
        const int rowbase = (it * 3 + iy) * 4 + idz;
#pragma unroll 1
        for (int iw = 0; iw < 3; ++iw) {
          const int t81 = ((it * 3 + iy) * 3 + idz) * 3 + iw;
          const int wp  = (wl + iw + 15) & 15;   // dw tap = rotation along N
#pragma unroll
          for (int cb = 0; cb < 2; ++cb) {
            const size_t abase = ((size_t)(t81 * 2 + cb) << 11) + aoff;
            v16bf Ah = ld_afrag(ahi + abase);
            v16bf Al = ld_afrag(alo + abase);
#pragma unroll
            for (int n = 0; n < 2; ++n) {
              int boff = ((((rowbase + n) * 2 + cb) << 4) + wp) * 32 + bk;
              v16bf Bh = ld_afrag(bhiS + boff);   // 16B at +0 and +32 bytes
              v16bf Bl = ld_afrag(bloS + boff);
              v8f& c = n ? acc1 : acc0;
              c = __builtin_amdgcn_wmma_f32_16x16x32_bf16(false, Ah, false, Bh,
                                                          (short)0, c, false, false);
              c = __builtin_amdgcn_wmma_f32_16x16x32_bf16(false, Ah, false, Bl,
                                                          (short)0, c, false, false);
              c = __builtin_amdgcn_wmma_f32_16x16x32_bf16(false, Al, false, Bh,
                                                          (short)0, c, false, false);
            }
          }
        }
      }

  // ---- epilogue: C/D layout VGPR r -> M = r + 8*(lane>=16), N = lane%16 ----
  const int cobase = (wave << 4) + ((lane >> 4) << 3);
#pragma unroll
  for (int n = 0; n < 2; ++n) {
    v8f c = n ? acc1 : acc0;
    int z = z0 + n;
#pragma unroll
    for (int r2 = 0; r2 < 8; ++r2) {
      int co = cobase + r2;
      size_t o = ((((((size_t)b * COUT + co) * S + t) * S + y) * S + z) << 4) + wl;
      out[o] = c[r2] + bias[co];
    }
  }
}

extern "C" void kernel_launch(void* const* d_in, const int* in_sizes, int n_in,
                              void* d_out, int out_size, void* d_ws, size_t ws_size,
                              hipStream_t stream) {
  const float* x    = (const float*)d_in[0];
  const float* w    = (const float*)d_in[1];
  const float* bias = (const float*)d_in[2];
  float* out = (float*)d_out;

  // workspace: packed bf16 weights, hi then lo (2 * 663552 B = 1.27 MB)
  bf16* ahi = (bf16*)d_ws;
  bf16* alo = ahi + APK_ELEMS;

  pack_w_kernel<<<(APK_ELEMS + 255) / 256, 256, 0, stream>>>(w, ahi, alo);

  const size_t smem = (size_t)2 * LDS_HALF * sizeof(bf16);  // 147456 B (<320KB/WGP)
  (void)hipFuncSetAttribute(reinterpret_cast<const void*>(conv4d_wmma_kernel),
                            hipFuncAttributeMaxDynamicSharedMemorySize, (int)smem);
  conv4d_wmma_kernel<<<4 * 16 * 16 * 8, 128, smem, stream>>>(x, ahi, alo, bias, out);
}
